// DirGNN_72756745994383
// MI455X (gfx1250) — matrix operand order repaired and checked by reading, not compiled, not run on hardware
//
#include <hip/hip_runtime.h>
#include <hip/hip_bf16.h>

// ---------------------------------------------------------------------------
// DirGNN (2-layer PNA, in/out edge channels) for MI455X / gfx1250 (wave32).
// bf16 WMMA GEMMs with f32 accumulation; CSR-based deterministic aggregation.
// ---------------------------------------------------------------------------

#define NNODES 20000
#define NEDGES 320000
#define FDIM   256
#define MAXD   96          // cached/sorted edges per node (deg ~ Poisson(16))
#define AVGL   2.8332133f  // log(17.0)

typedef __bf16 bf16x8 __attribute__((ext_vector_type(8)));
typedef __bf16 v16bf  __attribute__((ext_vector_type(16)));
typedef float  v8f    __attribute__((ext_vector_type(8)));

// ---- WMMA fragment loaders (CDNA5 ISA layouts, 16-bit operands) -----------
// A 16x32 (MxK): lane L -> row = L&15; elems[0..7]=K[8h..8h+7], elems[8..15]=K[16+8h..16+8h+7], h=L>>4
__device__ __forceinline__ v16bf load_frag_a(const __hip_bfloat16* base, size_t lda,
                                             int row0, int k0, int lane) {
  int h = lane >> 4;
  const __bf16* p = (const __bf16*)base + (size_t)(row0 + (lane & 15)) * lda + k0 + h * 8;
  bf16x8 lo = *(const bf16x8*)(p);
  bf16x8 hi = *(const bf16x8*)(p + 16);
  return __builtin_shufflevector(lo, hi, 0,1,2,3,4,5,6,7,8,9,10,11,12,13,14,15);
}
// B 32x16 (KxN) from row-major W[M,K]: lane L -> col = L&15; elems[0..7]=K[8h..8h+7], elems[8..15]=K[16+8h..16+8h+7]
__device__ __forceinline__ v16bf load_frag_b(const __hip_bfloat16* W, size_t ldb,
                                             int col0, int k0, int lane) {
  int h = lane >> 4;
  const __bf16* p = (const __bf16*)W + (size_t)(col0 + (lane & 15)) * ldb + k0 + h * 8;
  bf16x8 lo = *(const bf16x8*)(p);
  bf16x8 hi = *(const bf16x8*)(p + 16);
  return __builtin_shufflevector(lo, hi, 0,1,2,3,4,5,6,7,8,9,10,11,12,13,14,15);
}

// ---- Generic bf16 GEMM: C[N,M] = A[N,K] * W[M,K]^T (+bias)(+relu) ---------
// wave tile 16x64, 8 waves/block; M fixed = 256 via grid.y = 4.
__global__ __launch_bounds__(256) void gemm_kernel(
    const __hip_bfloat16* __restrict__ A, int lda,
    const __hip_bfloat16* __restrict__ B, int ldb,
    const float* __restrict__ bias,
    float* __restrict__ Cf, int ldcf,
    __hip_bfloat16* __restrict__ Cb, int ldcb,
    int nRowTiles, int K, int relu)
{
  int wave = threadIdx.x >> 5;
  int lane = threadIdx.x & 31;
  int rowTile = blockIdx.x * 8 + wave;
  if (rowTile >= nRowTiles) return;
  int row0 = rowTile * 16;
  int colBase = blockIdx.y * 64;

  v8f acc[4];
#pragma unroll
  for (int j = 0; j < 4; ++j)
#pragma unroll
    for (int i = 0; i < 8; ++i) acc[j][i] = 0.0f;

  for (int k0 = 0; k0 < K; k0 += 32) {
    v16bf a  = load_frag_a(A, (size_t)lda, row0, k0, lane);
    v16bf b0 = load_frag_b(B, (size_t)ldb, colBase +  0, k0, lane);
    v16bf b1 = load_frag_b(B, (size_t)ldb, colBase + 16, k0, lane);
    v16bf b2 = load_frag_b(B, (size_t)ldb, colBase + 32, k0, lane);
    v16bf b3 = load_frag_b(B, (size_t)ldb, colBase + 48, k0, lane);
    acc[0] = __builtin_amdgcn_wmma_f32_16x16x32_bf16(false, a, false, b0, (short)0, acc[0], false, false);
    acc[1] = __builtin_amdgcn_wmma_f32_16x16x32_bf16(false, a, false, b1, (short)0, acc[1], false, false);
    acc[2] = __builtin_amdgcn_wmma_f32_16x16x32_bf16(false, a, false, b2, (short)0, acc[2], false, false);
    acc[3] = __builtin_amdgcn_wmma_f32_16x16x32_bf16(false, a, false, b3, (short)0, acc[3], false, false);
  }

  int ln = lane & 15;
  int rh = (lane >> 4) * 8;
#pragma unroll
  for (int j = 0; j < 4; ++j) {
    int col = colBase + j * 16 + ln;
    float bv = bias ? bias[col] : 0.0f;
#pragma unroll
    for (int v = 0; v < 8; ++v) {
      float val = acc[j][v] + bv;
      if (relu) val = fmaxf(val, 0.0f);
      size_t row = (size_t)(row0 + rh + v);
      if (Cf) Cf[row * (size_t)ldcf + col] = val;
      if (Cb) Cb[row * (size_t)ldcb + col] = __float2bfloat16(val);
    }
  }
}

// ---- CSR build ------------------------------------------------------------
__global__ void deg_kernel(const int* __restrict__ dst, int* __restrict__ deg, int E) {
  int e = blockIdx.x * blockDim.x + threadIdx.x;
  if (e < E) atomicAdd(&deg[dst[e]], 1);
}

__global__ __launch_bounds__(256) void scan_kernel(const int* __restrict__ deg,
                                                   int* __restrict__ rowstart, int n) {
  __shared__ int sh[256];
  int t = threadIdx.x;
  int chunk = (n + 255) / 256;
  int b = t * chunk;
  int e = b + chunk; if (e > n) e = n;
  int ssum = 0;
  for (int i = b; i < e; ++i) ssum += deg[i];
  sh[t] = ssum;
  __syncthreads();
  for (int off = 1; off < 256; off <<= 1) {
    int v = sh[t];
    int u = (t >= off) ? sh[t - off] : 0;
    __syncthreads();
    sh[t] = v + u;
    __syncthreads();
  }
  int run = (t == 0) ? 0 : sh[t - 1];
  for (int i = b; i < e; ++i) { rowstart[i] = run; run += deg[i]; }
}

__global__ void scatter_kernel(const int* __restrict__ dst, const int* __restrict__ rowstart,
                               int* __restrict__ cursor, int* __restrict__ elist, int E) {
  int e = blockIdx.x * blockDim.x + threadIdx.x;
  if (e < E) {
    int d = dst[e];
    int pos = rowstart[d] + atomicAdd(&cursor[d], 1);
    elist[pos] = e;
  }
}

// ---- Per-node aggregation + PNA finalize into bf16 [N,4096] ---------------
// One wave per node; edge list sorted (canonical, deterministic fp order).
__global__ __launch_bounds__(256) void aggregate_kernel(
    const int* __restrict__ srcArr,
    const int* __restrict__ rowstart, const int* __restrict__ degArr,
    const int* __restrict__ elist,
    const float* __restrict__ P,      // [N,512] = pd(+bpre) | ps
    const float* __restrict__ x,      // [N,256] current features (f32)
    __hip_bfloat16* __restrict__ AGG, // [N,4096] post-NN input
    int N)
{
  __shared__ int sh[8 * MAXD];
  int wave = threadIdx.x >> 5, lane = threadIdx.x & 31;
  int node = blockIdx.x * 8 + wave;
  bool active = node < N;
  int start = 0, cnt = 0;
  if (active) { start = rowstart[node]; cnt = degArr[node]; }
  int cached = (cnt < MAXD) ? cnt : MAXD;
  int* my = sh + wave * MAXD;
  for (int i = lane; i < cached; i += 32) my[i] = elist[start + i];
  __syncthreads();
  if (active && lane == 0) {                 // insertion sort -> canonical order
    for (int i = 1; i < cached; ++i) {
      int v = my[i]; int j = i - 1;
      while (j >= 0 && my[j] > v) { my[j + 1] = my[j]; --j; }
      my[j + 1] = v;
    }
  }
  __syncthreads();
  if (!active) return;

  float pd[8], s[8], s2[8], mn[8], mx[8];
  const float* pdr = P + (size_t)node * 512;
#pragma unroll
  for (int u = 0; u < 8; ++u) {
    pd[u] = pdr[lane + 32 * u];
    s[u] = 0.0f; s2[u] = 0.0f;
    mn[u] = __builtin_inff(); mx[u] = -__builtin_inff();
  }
  for (int i = 0; i < cnt; ++i) {
    int e = (i < cached) ? my[i] : elist[start + i];
    int sn = srcArr[e];
    const float* psr = P + (size_t)sn * 512 + 256;
#pragma unroll
    for (int u = 0; u < 8; ++u) {
      float m = pd[u] + psr[lane + 32 * u];
      s[u] += m; s2[u] += m * m;
      mn[u] = fminf(mn[u], m); mx[u] = fmaxf(mx[u], m);
    }
  }
  float degc = fmaxf((float)cnt, 1.0f);
  float inv = 1.0f / degc;
  float logd = logf(degc + 1.0f);
  float sc1 = logd * (1.0f / AVGL);
  float sc2 = AVGL / logd;
  __hip_bfloat16* row = AGG + (size_t)node * 4096;
#pragma unroll
  for (int u = 0; u < 8; ++u) {
    int f = lane + 32 * u;
    row[f] = __float2bfloat16(x[(size_t)node * 256 + f]);
    float mean = s[u] * inv;
    float var = fmaxf(s2[u] * inv - mean * mean, 0.0f);
    float sd = sqrtf(var + 1e-5f);
    float mnv = (cnt > 0) ? mn[u] : 0.0f;
    float mxv = (cnt > 0) ? mx[u] : 0.0f;
    float ag[5] = { mean, s[u], sd, mnv, mxv };
#pragma unroll
    for (int a = 0; a < 5; ++a) {
      row[(size_t)256 * (1 + a)  + f] = __float2bfloat16(ag[a]);
      row[(size_t)256 * (6 + a)  + f] = __float2bfloat16(ag[a] * sc1);
      row[(size_t)256 * (11 + a) + f] = __float2bfloat16(ag[a] * sc2);
    }
  }
}

// ---- Small helpers --------------------------------------------------------
__global__ void cvt_kernel(const float* __restrict__ s, __hip_bfloat16* __restrict__ d, int n) {
  int i = blockIdx.x * blockDim.x + threadIdx.x;
  int stride = gridDim.x * blockDim.x;
  for (; i < n; i += stride) d[i] = __float2bfloat16(s[i]);
}

__global__ void xcopy_kernel(const float* __restrict__ x, __hip_bfloat16* __restrict__ cb) {
  size_t n = blockIdx.x; int t = threadIdx.x;
  cb[n * 768 + t] = __float2bfloat16(x[n * 256 + t]);
}

__global__ void out_kernel(const float* __restrict__ x, const float* __restrict__ Wout,
                           const float* __restrict__ bout, float* __restrict__ out, int N) {
  int wid = (blockIdx.x * blockDim.x + threadIdx.x) >> 5;
  int lane = threadIdx.x & 31;
  if (wid >= N) return;
  const float* xr = x + (size_t)wid * 256;
  float acc = 0.0f;
#pragma unroll
  for (int u = 0; u < 8; ++u) acc += xr[lane + 32 * u] * Wout[lane + 32 * u];
  for (int off = 16; off; off >>= 1) acc += __shfl_down(acc, off);
  if (lane == 0) out[wid] = acc + bout[0];
}

// ---------------------------------------------------------------------------
extern "C" void kernel_launch(void* const* d_in, const int* in_sizes, int n_in,
                              void* d_out, int out_size, void* d_ws, size_t ws_size,
                              hipStream_t stream) {
  (void)in_sizes; (void)n_in; (void)out_size; (void)ws_size;
  const int N = NNODES, E = NEDGES;

  const float* x_in   = (const float*)d_in[0];
  const int*   ei_in  = (const int*)d_in[1];
  const int*   ei_out = (const int*)d_in[2];
  const float* Wpre_s = (const float*)d_in[3];
  const float* bpre_s = (const float*)d_in[4];
  const float* Wpost_s= (const float*)d_in[5];
  const float* bpost_s= (const float*)d_in[6];
  const float* Wlin_s = (const float*)d_in[7];
  const float* blin_s = (const float*)d_in[8];
  const float* Wpre_d = (const float*)d_in[9];
  const float* bpre_d = (const float*)d_in[10];
  const float* Wpost_d= (const float*)d_in[11];
  const float* bpost_d= (const float*)d_in[12];
  const float* Wlin_d = (const float*)d_in[13];
  const float* blin_d = (const float*)d_in[14];
  const float* Wcomb  = (const float*)d_in[15];
  const float* bcomb  = (const float*)d_in[16];
  const float* Wout   = (const float*)d_in[17];
  const float* bout   = (const float*)d_in[18];

  // ---- carve workspace ----
  char* p = (char*)d_ws;
  auto alloc = [&](size_t bytes) -> void* {
    void* r = (void*)p; p += (bytes + 255) & ~(size_t)255; return r;
  };
  __hip_bfloat16* Wpre_s_b  = (__hip_bfloat16*)alloc((size_t)2*256*512*2);
  __hip_bfloat16* Wpre_d_b  = (__hip_bfloat16*)alloc((size_t)2*256*512*2);
  __hip_bfloat16* Wpost_s_b = (__hip_bfloat16*)alloc((size_t)2*256*4096*2);
  __hip_bfloat16* Wpost_d_b = (__hip_bfloat16*)alloc((size_t)2*256*4096*2);
  __hip_bfloat16* Wlin_s_b  = (__hip_bfloat16*)alloc((size_t)2*256*256*2);
  __hip_bfloat16* Wlin_d_b  = (__hip_bfloat16*)alloc((size_t)2*256*256*2);
  __hip_bfloat16* Wcomb_b   = (__hip_bfloat16*)alloc((size_t)2*256*768*2);
  __hip_bfloat16* CB0 = (__hip_bfloat16*)alloc((size_t)N*768*2);   // [x|h_in|h_out] bf16
  __hip_bfloat16* CB1 = (__hip_bfloat16*)alloc((size_t)N*768*2);
  float*          xf  = (float*)alloc((size_t)N*256*4);            // layer output f32
  float*          P   = (float*)alloc((size_t)N*512*4);            // pd|ps
  __hip_bfloat16* AGG = (__hip_bfloat16*)alloc((size_t)N*4096*2);  // post-NN input
  __hip_bfloat16* Tb  = (__hip_bfloat16*)alloc((size_t)N*256*2);   // post-NN output
  int* degA = (int*)alloc((size_t)N*4);
  int* rsA  = (int*)alloc((size_t)N*4);
  int* curA = (int*)alloc((size_t)N*4);
  int* elA  = (int*)alloc((size_t)E*4);
  int* degB = (int*)alloc((size_t)N*4);
  int* rsB  = (int*)alloc((size_t)N*4);
  int* curB = (int*)alloc((size_t)N*4);
  int* elB  = (int*)alloc((size_t)E*4);

  const int nRowTiles = N / 16;                 // 1250
  const dim3 ggrid((nRowTiles + 7) / 8, 4);     // (157, 4), M = 256
  auto gemm = [&](const __hip_bfloat16* A, int lda, const __hip_bfloat16* B, int ldb,
                  const float* bias, float* Cf, int ldcf, __hip_bfloat16* Cb, int ldcb,
                  int K, int relu) {
    gemm_kernel<<<ggrid, 256, 0, stream>>>(A, lda, B, ldb, bias, Cf, ldcf, Cb, ldcb,
                                           nRowTiles, K, relu);
  };

  // ---- bf16 weight conversions (each call; deterministic, cheap) ----
  auto cvt = [&](const float* s, __hip_bfloat16* d, int n) {
    cvt_kernel<<<1024, 256, 0, stream>>>(s, d, n);
  };
  cvt(Wpre_s, Wpre_s_b, 2*256*512);
  cvt(Wpre_d, Wpre_d_b, 2*256*512);
  cvt(Wpost_s, Wpost_s_b, 2*256*4096);
  cvt(Wpost_d, Wpost_d_b, 2*256*4096);
  cvt(Wlin_s, Wlin_s_b, 2*256*256);
  cvt(Wlin_d, Wlin_d_b, 2*256*256);
  cvt(Wcomb, Wcomb_b, 2*256*768);

  // ---- CSR for both edge sets (src -> dst; aggregate at dst) ----
  auto buildCSR = [&](const int* edges, int* dg, int* rs, int* cur, int* el) {
    const int* dst = edges + E;                 // edge_index[1]
    hipMemsetAsync(dg, 0, (size_t)N*4, stream);
    hipMemsetAsync(cur, 0, (size_t)N*4, stream);
    deg_kernel<<<(E + 255) / 256, 256, 0, stream>>>(dst, dg, E);
    scan_kernel<<<1, 256, 0, stream>>>(dg, rs, N);
    scatter_kernel<<<(E + 255) / 256, 256, 0, stream>>>(dst, rs, cur, el, E);
  };
  buildCSR(ei_in, degA, rsA, curA, elA);
  buildCSR(ei_out, degB, rsB, curB, elB);

  // seed bf16 x-block of CB0 from input x
  xcopy_kernel<<<N, 256, 0, stream>>>(x_in, CB0);

  // ---- one PNA conv: pre-GEMMs -> aggregate -> post GEMM -> lin GEMM ----
  auto conv = [&](const int* edges, const int* rs, const int* dg, const int* el,
                  const __hip_bfloat16* WpreB, const float* bpre,
                  const __hip_bfloat16* WpostB, const float* bpost,
                  const __hip_bfloat16* WlinB, const float* blin,
                  const __hip_bfloat16* CB, const float* xcur, __hip_bfloat16* hOut) {
    const int* srcArr = edges;                  // edge_index[0]
    // pd = x@Wl.T + bpre ;  ps = x@Wr.T   (Wpre = [Wl | Wr], row-major [256,512])
    gemm(CB, 768, WpreB,       512, bpre,    P,       512, nullptr, 0, 256, 0);
    gemm(CB, 768, WpreB + 256, 512, nullptr, P + 256, 512, nullptr, 0, 256, 0);
    aggregate_kernel<<<(N + 7) / 8, 256, 0, stream>>>(srcArr, rs, dg, el, P, xcur, AGG, N);
    gemm(AGG, 4096, WpostB, 4096, bpost, nullptr, 0, Tb,   256, 4096, 0);
    gemm(Tb,  256,  WlinB,  256,  blin,  nullptr, 0, hOut, 768, 256, 1);   // relu
  };

  // ---- layer 0 ----
  conv(ei_in,  rsA, degA, elA, Wpre_s_b, bpre_s, Wpost_s_b, bpost_s, Wlin_s_b, blin_s,
       CB0, x_in, CB0 + 256);
  conv(ei_out, rsB, degB, elB, Wpre_d_b, bpre_d, Wpost_d_b, bpost_d, Wlin_d_b, blin_d,
       CB0, x_in, CB0 + 512);
  // x1 = relu([x|h_in|h_out]@Wcomb.T + bcomb) -> xf (f32) and CB1 x-block (bf16)
  gemm(CB0, 768, Wcomb_b, 768, bcomb, xf, 256, CB1, 768, 768, 1);

  // ---- layer 1 ----
  conv(ei_in,  rsA, degA, elA, Wpre_s_b + 2*256*256, bpre_s + 256,
       Wpost_s_b + (size_t)256*4096, bpost_s + 256, Wlin_s_b + 256*256, blin_s + 256,
       CB1, xf, CB1 + 256);
  conv(ei_out, rsB, degB, elB, Wpre_d_b + 2*256*256, bpre_d + 256,
       Wpost_d_b + (size_t)256*4096, bpost_d + 256, Wlin_d_b + 256*256, blin_d + 256,
       CB1, xf, CB1 + 512);
  gemm(CB1, 768, Wcomb_b + 256*768, 768, bcomb + 256, xf, 256, nullptr, 0, 768, 1);

  // ---- readout: out[n] = x2[n] . Wout + bout ----
  out_kernel<<<(N * 32 + 255) / 256, 256, 0, stream>>>(xf, Wout, bout, (float*)d_out, N);
}
